// WindowAttentionSDPA_4114578670394
// MI455X (gfx1250) — compile-verified, matrix operand order checked by reading, not run on hardware
//
#include <hip/hip_runtime.h>

// ---------------------------------------------------------------------------
// Swin-V2 window attention block for MI455X (gfx1250), wave32 + WMMA bf16.
// All GEMMs: v_wmma_f32_16x16x32_bf16 (fp32 accumulate).
// GEMM waves are register-tiled 16x64 (4 accumulators, A-fragment reuse);
// weights pre-converted to bf16 so every fragment is 2x global_load_b128.
// ---------------------------------------------------------------------------

typedef __bf16 bf16_t;
typedef __attribute__((ext_vector_type(16))) __bf16 v16bf;
typedef __attribute__((ext_vector_type(8)))  __bf16 v8bf;
typedef __attribute__((ext_vector_type(8)))  float  v8f;

#define DIMC   256
#define HEADS  8
#define HD     32
#define WS     8
#define NTOK   64
#define BATCH  16
#define HW     4096      // 64*64
#define NPOS   65536     // BATCH*HW

// ---- WMMA fragment loader (16-bit A/B layout, ISA 7.12.2) -----------------
// Per lane: 8 contiguous bf16 at +0 and 8 contiguous bf16 at +16 elements.
// Caller passes pointer already offset by (lane>>4)*8 within the K run.
__device__ __forceinline__ v16bf frag_ld_bf16(const bf16_t* p) {
    v8bf lo = *(const v8bf*)(p);
    v8bf hi = *(const v8bf*)(p + 16);
    return __builtin_shufflevector(lo, hi, 0,1,2,3,4,5,6,7,8,9,10,11,12,13,14,15);
}

// ---------------------------------------------------------------------------
// Kernel 0: fp32 -> bf16 weight conversion (one-time, ~1 MB total)
// ---------------------------------------------------------------------------
__global__ void __launch_bounds__(256)
k_cvt_bf16(const float* __restrict__ src, bf16_t* __restrict__ dst, int n) {
    int i = blockIdx.x * 256 + threadIdx.x;
    if (i < n) dst[i] = (bf16_t)src[i];
}

// ---------------------------------------------------------------------------
// Kernel 1: continuous-position-bias table -> bias[8][64][64]
// ---------------------------------------------------------------------------
__global__ void __launch_bounds__(256)
k_bias(const float* __restrict__ w1, const float* __restrict__ b1,
       const float* __restrict__ w2, float* __restrict__ bias) {
    __shared__ float btab[225][8];
    const float BN = 0.9999950000374997f;   // 1/sqrt(1+1e-5)
    int t = threadIdx.x;
    if (t < 225) {
        int i = t / 15, j = t % 15;
        float fi = (float)(i - 7) * (8.0f / 7.0f);
        float fj = (float)(j - 7) * (8.0f / 7.0f);
        float si = (fi > 0.f) ? 1.f : ((fi < 0.f) ? -1.f : 0.f);
        float sj = (fj > 0.f) ? 1.f : ((fj < 0.f) ? -1.f : 0.f);
        float gi = si * __log2f(fabsf(fi) + 1.f) * (1.0f / 3.0f);
        float gj = sj * __log2f(fabsf(fj) + 1.f) * (1.0f / 3.0f);
        float acc[8];
#pragma unroll
        for (int h = 0; h < 8; ++h) acc[h] = 0.f;
        for (int d = 0; d < DIMC; ++d) {
            float pre = (gi * w1[d * 2] + gj * w1[d * 2 + 1] + b1[d]) * BN;
            float hs  = pre / (1.f + __expf(-pre));   // SiLU
#pragma unroll
            for (int h = 0; h < 8; ++h) acc[h] += hs * w2[h * DIMC + d];
        }
#pragma unroll
        for (int h = 0; h < 8; ++h) btab[t][h] = acc[h] * BN;
    }
    __syncthreads();
    for (int idx = threadIdx.x; idx < HEADS * NTOK * NTOK; idx += 256) {
        int h = idx >> 12;
        int q = (idx >> 6) & 63;
        int k = idx & 63;
        int dy = (q >> 3) - (k >> 3) + 7;
        int dx = (q & 7)  - (k & 7)  + 7;
        float v = btab[dy * 15 + dx][h];
        bias[idx] = 16.f / (1.f + __expf(-v));
    }
}

// ---------------------------------------------------------------------------
// Kernel 2: x (B,C,H,W) fp32 -> xb[p][c] bf16 (channel-contiguous)
// 32x32 LDS-tiled transpose, block = 32x8.
// ---------------------------------------------------------------------------
__global__ void __launch_bounds__(256)
k_transpose(const float* __restrict__ x, bf16_t* __restrict__ xb) {
    __shared__ float t[32][33];
    int b   = blockIdx.z;
    int ct  = blockIdx.y;      // channel tile (8)
    int hwt = blockIdx.x;      // spatial tile (128)
    int tx = threadIdx.x & 31, ty = threadIdx.x >> 5;
    const float* xp = x + ((size_t)b * DIMC + ct * 32) * HW + hwt * 32;
#pragma unroll
    for (int i = 0; i < 4; ++i) {
        int c = ty + i * 8;
        t[c][tx] = xp[(size_t)c * HW + tx];
    }
    __syncthreads();
    bf16_t* op = xb + ((size_t)b * HW + hwt * 32) * DIMC + ct * 32;
#pragma unroll
    for (int i = 0; i < 4; ++i) {
        int hw = ty + i * 8;
        op[(size_t)hw * DIMC + tx] = (bf16_t)t[tx][hw];
    }
}

// ---------------------------------------------------------------------------
// Kernel 3: QKV GEMM: qkvb[p][o] = sum_c xb[p][c] * qkv_wb[o][c] + qkv_b[o]
// M=65536 (p), N=768 (o), K=256. Wave computes 16x64 (4 acc tiles, A reuse).
// 49152 tiles of 16x64 -> 6144 blocks x 8 waves.
// ---------------------------------------------------------------------------
__global__ void __launch_bounds__(256)
k_qkv_gemm(const bf16_t* __restrict__ xb, const bf16_t* __restrict__ wb,
           const float* __restrict__ bvec, bf16_t* __restrict__ qkvb) {
    int wid = threadIdx.x >> 5, lane = threadIdx.x & 31;
    int tile = blockIdx.x * 8 + wid;
    int pt = tile / 12, os = tile % 12;       // os: 64-wide output supertile
    int m16 = lane & 15, kh8 = (lane >> 4) * 8;
    const bf16_t* ap = xb + (size_t)(pt * 16 + m16) * DIMC + kh8;
    const bf16_t* bp = wb + (size_t)(os * 64 + m16) * DIMC + kh8;
    v8f acc[4] = {{}, {}, {}, {}};
#pragma unroll
    for (int kt = 0; kt < 8; ++kt) {
        v16bf a = frag_ld_bf16(ap + kt * 32);
#pragma unroll
        for (int s = 0; s < 4; ++s) {
            v16bf b = frag_ld_bf16(bp + (size_t)s * 16 * DIMC + kt * 32);
            acc[s] = __builtin_amdgcn_wmma_f32_16x16x32_bf16(
                         false, a, false, b, (short)0, acc[s], false, false);
        }
    }
#pragma unroll
    for (int s = 0; s < 4; ++s) {
        int o = os * 64 + s * 16 + m16;
        float bb = bvec[o];
#pragma unroll
        for (int r = 0; r < 8; ++r) {
            int p = pt * 16 + (lane >> 4) * 8 + r;
            qkvb[(size_t)p * 768 + o] = (bf16_t)(acc[s][r] + bb);
        }
    }
}

// ---------------------------------------------------------------------------
// Kernel 4: windowed SDPA. One block (4 waves) per (window, head).
// scores = Q K^T / sqrt(32) + bias[h]; softmax; O = P V.
// ---------------------------------------------------------------------------
__global__ void __launch_bounds__(128)
k_attention(const bf16_t* __restrict__ qkvb, const float* __restrict__ bias,
            bf16_t* __restrict__ ob) {
    __shared__ float  sc[64][64];   // 16 KB scores
    __shared__ bf16_t pb[64][64];   //  8 KB probs (bf16 for WMMA)
    __shared__ bf16_t vt[32][64];   //  4 KB V^T
    int wid = threadIdx.x >> 5, lane = threadIdx.x & 31;
    int win = blockIdx.x >> 3, head = blockIdx.x & 7;
    int b = win >> 6, rem = win & 63, hn = rem >> 3, wn = rem & 7;
    int p_base = b * HW + hn * 8 * 64 + wn * 8;
#define TOKP(t) (p_base + ((t) >> 3) * 64 + ((t) & 7))

    // stage V^T into LDS (coalesced-ish read, transposed write)
    {
        int t  = threadIdx.x >> 1;
        int d0 = (threadIdx.x & 1) * 16;
        const bf16_t* vp = qkvb + (size_t)TOKP(t) * 768 + 512 + head * HD + d0;
#pragma unroll
        for (int i = 0; i < 16; ++i) vt[d0 + i][t] = vp[i];
    }

    int m16 = lane & 15, kh8 = (lane >> 4) * 8;

    // scores: wave wid handles query tile wid, loops over 4 key tiles (K=32 -> 1 WMMA)
    int tq = wid * 16 + m16;
    v16bf aq = frag_ld_bf16(qkvb + (size_t)TOKP(tq) * 768 + head * HD + kh8);
    const float* bh = bias + head * NTOK * NTOK;
#pragma unroll
    for (int nt = 0; nt < 4; ++nt) {
        int tk = nt * 16 + m16;
        v16bf bk = frag_ld_bf16(qkvb + (size_t)TOKP(tk) * 768 + 256 + head * HD + kh8);
        v8f c = {};
        c = __builtin_amdgcn_wmma_f32_16x16x32_bf16(false, aq, false, bk,
                                                    (short)0, c, false, false);
#pragma unroll
        for (int r = 0; r < 8; ++r) {
            int m = wid * 16 + (lane >> 4) * 8 + r;
            int n = nt * 16 + m16;
            sc[m][n] = c[r] * 0.17677669529663687f + bh[m * 64 + n];
        }
    }
    __syncthreads();

    // softmax: one thread per row
    if (threadIdx.x < 64) {
        int row = threadIdx.x;
        float mx = -1e30f;
#pragma unroll 8
        for (int j = 0; j < 64; ++j) mx = fmaxf(mx, sc[row][j]);
        float s = 0.f;
#pragma unroll 8
        for (int j = 0; j < 64; ++j) { float e = __expf(sc[row][j] - mx); sc[row][j] = e; s += e; }
        float inv = 1.f / s;
#pragma unroll 8
        for (int j = 0; j < 64; ++j) pb[row][j] = (bf16_t)(sc[row][j] * inv);
    }
    __syncthreads();

    // O = P (64x64) @ V (64x32): wave wid -> row tile wid, 2 col tiles, 2 K-steps
#pragma unroll
    for (int nt = 0; nt < 2; ++nt) {
        v8f c = {};
#pragma unroll
        for (int ks = 0; ks < 2; ++ks) {
            v16bf a  = frag_ld_bf16(&pb[wid * 16 + m16][ks * 32 + kh8]);
            v16bf bv = frag_ld_bf16(&vt[nt * 16 + m16][ks * 32 + kh8]);
            c = __builtin_amdgcn_wmma_f32_16x16x32_bf16(false, a, false, bv,
                                                        (short)0, c, false, false);
        }
#pragma unroll
        for (int r = 0; r < 8; ++r) {
            int m = wid * 16 + (lane >> 4) * 8 + r;
            int d = nt * 16 + m16;
            ob[(size_t)TOKP(m) * DIMC + head * HD + d] = (bf16_t)c[r];
        }
    }
#undef TOKP
}

// ---------------------------------------------------------------------------
// Kernel 5: depthwise 7x7 conv on V (read straight from qkvb; window reverse
// of V is the identity map) + residual add of attention output -> pin (bf16)
// ---------------------------------------------------------------------------
__global__ void __launch_bounds__(64)
k_dwconv_add(const bf16_t* __restrict__ qkvb, const bf16_t* __restrict__ ob,
             const float* __restrict__ dww, bf16_t* __restrict__ pin) {
    int bid = blockIdx.x;
    int b = bid >> 14;          // / (256*64)
    int c = (bid >> 6) & 255;
    int h = bid & 63;
    int w = threadIdx.x;
    const float* wp = dww + c * 49;
    float acc = 0.f;
#pragma unroll
    for (int ky = 0; ky < 7; ++ky) {
        int y = h + ky - 3;
        if ((unsigned)y >= 64u) continue;
#pragma unroll
        for (int kx = 0; kx < 7; ++kx) {
            int xx = w + kx - 3;
            if ((unsigned)xx >= 64u) continue;
            size_t p = (size_t)b * HW + y * 64 + xx;
            acc += wp[ky * 7 + kx] * (float)qkvb[p * 768 + 512 + c];
        }
    }
    size_t p = (size_t)b * HW + h * 64 + w;
    pin[p * DIMC + c] = (bf16_t)((float)ob[p * DIMC + c] + acc);
}

// ---------------------------------------------------------------------------
// Kernel 6: proj GEMM: out[b][o][hw] = sum_c pin[p][c]*proj_wb[o][c] + proj_b[o]
// M=65536 (p), N=256 (o), K=256. Wave computes 16x64. fp32 NCHW output.
// 16384 tiles -> 2048 blocks x 8 waves.
// ---------------------------------------------------------------------------
__global__ void __launch_bounds__(256)
k_proj_gemm(const bf16_t* __restrict__ pin, const bf16_t* __restrict__ wb,
            const float* __restrict__ bvec, float* __restrict__ out) {
    int wid = threadIdx.x >> 5, lane = threadIdx.x & 31;
    int tile = blockIdx.x * 8 + wid;
    int pt = tile >> 2, os = tile & 3;        // os: 64-wide output supertile
    int m16 = lane & 15, kh8 = (lane >> 4) * 8;
    const bf16_t* ap = pin + (size_t)(pt * 16 + m16) * DIMC + kh8;
    const bf16_t* bp = wb  + (size_t)(os * 64 + m16) * DIMC + kh8;
    v8f acc[4] = {{}, {}, {}, {}};
#pragma unroll
    for (int kt = 0; kt < 8; ++kt) {
        v16bf a = frag_ld_bf16(ap + kt * 32);
#pragma unroll
        for (int s = 0; s < 4; ++s) {
            v16bf b = frag_ld_bf16(bp + (size_t)s * 16 * DIMC + kt * 32);
            acc[s] = __builtin_amdgcn_wmma_f32_16x16x32_bf16(
                         false, a, false, b, (short)0, acc[s], false, false);
        }
    }
#pragma unroll
    for (int s = 0; s < 4; ++s) {
        int o = os * 64 + s * 16 + m16;
        float bb = bvec[o];
#pragma unroll
        for (int r = 0; r < 8; ++r) {
            int p  = pt * 16 + (lane >> 4) * 8 + r;
            int bi = p >> 12;          // batch
            int hw = p & 4095;
            out[(size_t)bi * (DIMC * HW) + (size_t)o * HW + hw] = acc[s][r] + bb;
        }
    }
}

// ---------------------------------------------------------------------------
extern "C" void kernel_launch(void* const* d_in, const int* in_sizes, int n_in,
                              void* d_out, int out_size, void* d_ws, size_t ws_size,
                              hipStream_t stream) {
    const float* x      = (const float*)d_in[0];
    const float* qkv_w  = (const float*)d_in[1];
    const float* qkv_b  = (const float*)d_in[2];
    const float* proj_w = (const float*)d_in[3];
    const float* proj_b = (const float*)d_in[4];
    const float* cpb_w1 = (const float*)d_in[5];
    const float* cpb_b1 = (const float*)d_in[6];
    const float* cpb_w2 = (const float*)d_in[7];
    const float* dw_w   = (const float*)d_in[8];
    float* out = (float*)d_out;

    // workspace layout (bytes, 256-aligned)
    char* ws = (char*)d_ws;
    size_t off = 0;
    float*  bias    = (float*)(ws + off);  off += 131072;            // 8*64*64*4
    bf16_t* qkv_wb  = (bf16_t*)(ws + off); off += 393216;            // 768*256*2
    bf16_t* proj_wb = (bf16_t*)(ws + off); off += 131072;            // 256*256*2
    bf16_t* xb      = (bf16_t*)(ws + off); off += 33554432;          // 65536*256*2
    bf16_t* qkvb    = (bf16_t*)(ws + off); off += 100663296;         // 65536*768*2
    bf16_t* ob      = (bf16_t*)(ws + off); off += 33554432;          // 65536*256*2
    bf16_t* pin     = xb;   // xb is dead after k_qkv_gemm; reuse as proj input

    k_cvt_bf16  <<<768, 256, 0, stream>>>(qkv_w, qkv_wb, 768 * 256);
    k_cvt_bf16  <<<256, 256, 0, stream>>>(proj_w, proj_wb, 256 * 256);
    k_bias      <<<1, 256, 0, stream>>>(cpb_w1, cpb_b1, cpb_w2, bias);
    k_transpose <<<dim3(128, 8, 16), 256, 0, stream>>>(x, xb);
    k_qkv_gemm  <<<6144, 256, 0, stream>>>(xb, qkv_wb, qkv_b, qkvb);
    k_attention <<<8192, 128, 0, stream>>>(qkvb, bias, ob);
    k_dwconv_add<<<262144, 64, 0, stream>>>(qkvb, ob, dw_w, pin);
    k_proj_gemm <<<2048, 256, 0, stream>>>(pin, proj_wb, proj_b, out);
}
